// DotProductAttention_79946521247837
// MI455X (gfx1250) — matrix operand order repaired and checked by reading, not compiled, not run on hardware
//
#include <hip/hip_runtime.h>
#include <hip/hip_bf16.h>
#include <math.h>

typedef __attribute__((ext_vector_type(16))) _Float16 v16h;
typedef __attribute__((ext_vector_type(4)))  _Float16 v4h;
typedef __attribute__((ext_vector_type(8)))  float    v8f;
typedef __attribute__((ext_vector_type(4)))  float    v4f;

#define SEQ   4096
#define DIM   64
#define KT    64      // keys staged per tile
#define WAVES 8       // waves per workgroup (wave32)
#define QW    16      // query rows per wave

// A-fragment K offset, contiguous k = 16*kh + s (pair base for VGPR v, lane-half kh)
__device__ __forceinline__ int a_kbase(int v, int kh) {
    return 2 * v + 16 * kh;
}

__global__ __launch_bounds__(256)
void fa_masked_sdpa_kernel(const float* __restrict__ Q,
                           const float* __restrict__ Km,
                           const float* __restrict__ Vm,
                           const int*   __restrict__ VL,
                           float* __restrict__ Out)
{
    __shared__ _Float16 Klds[KT * DIM];        // [key][dim] row-major f16 (8 KB)
    __shared__ _Float16 VTlds[DIM * KT];       // [dim][key] transposed f16 (8 KB)
    __shared__ _Float16 Plds[WAVES][QW * KT];  // per-wave P tile [m][col] (16 KB)

    const int tid  = threadIdx.x;
    const int wave = tid >> 5;
    const int lane = tid & 31;
    const int ln   = lane & 15;   // A row / B,C column
    const int kh   = lane >> 4;   // K-half selector

    const int b      = blockIdx.x >> 5;   // 32 workgroups per batch
    const int qblock = blockIdx.x & 31;
    const int qbase  = qblock * (WAVES * QW) + wave * QW;

    const int vlen   = VL[b];
    const int ntiles = (vlen + KT - 1) / KT;   // skip fully-masked key tiles (exact)

    const size_t bOff = (size_t)b * SEQ * DIM;

    // ---- Q -> two A fragments (16x32 f16 each), contiguous k = 16*kh + s ----
    v16h qa[2];
    {
        const float* qrow = Q + bOff + (size_t)(qbase + ln) * DIM;
#pragma unroll
        for (int c = 0; c < 2; ++c) {
#pragma unroll
            for (int v = 0; v < 8; ++v) {
                const int kb = a_kbase(v, kh) + 32 * c;
                qa[c][2 * v + 0] = (_Float16)qrow[kb + 0];
                qa[c][2 * v + 1] = (_Float16)qrow[kb + 1];
            }
        }
    }

    // streaming softmax state: row m = v + 8*kh lives at VGPR index v of this lane
    float mrow[8], lrow[8];
    v8f   oacc[4];
#pragma unroll
    for (int v = 0; v < 8; ++v) { mrow[v] = -INFINITY; lrow[v] = 0.0f; }
#pragma unroll
    for (int c = 0; c < 4; ++c) oacc[c] = (v8f){};

    for (int t = 0; t < ntiles; ++t) {
        __syncthreads();   // previous tile fully consumed by all waves

        // ---- cooperative stage: K row-major, V transposed, fp32->f16 ----
#pragma unroll
        for (int r = 0; r < 4; ++r) {
            const int idx = tid + 256 * r;     // 0..1023 float4 chunks
            const int key = idx >> 4;
            const int j4  = (idx & 15) * 4;
            const size_t g = bOff + (size_t)(t * KT + key) * DIM + j4;

            const v4f kv = *(const v4f*)(Km + g);
            v4h k4;
            k4[0] = (_Float16)kv[0]; k4[1] = (_Float16)kv[1];
            k4[2] = (_Float16)kv[2]; k4[3] = (_Float16)kv[3];
            *(v4h*)&Klds[key * DIM + j4] = k4;

            const v4f vv = *(const v4f*)(Vm + g);
            VTlds[(j4 + 0) * KT + key] = (_Float16)vv[0];
            VTlds[(j4 + 1) * KT + key] = (_Float16)vv[1];
            VTlds[(j4 + 2) * KT + key] = (_Float16)vv[2];
            VTlds[(j4 + 3) * KT + key] = (_Float16)vv[3];
        }
        __syncthreads();

        // ---- scores: S = Q * K^T for 16 rows x 64 keys (8 WMMAs) ----
        v8f sc[4];
#pragma unroll
        for (int n = 0; n < 4; ++n) {
            v8f c = (v8f){};
            const v16h bk0 = *(const v16h*)&Klds[(n * 16 + ln) * DIM + 16 * kh];
            const v16h bk1 = *(const v16h*)&Klds[(n * 16 + ln) * DIM + 32 + 16 * kh];
            c = __builtin_amdgcn_wmma_f32_16x16x32_f16(false, qa[0], false, bk0,
                                                       (short)0, c, false, false);
            c = __builtin_amdgcn_wmma_f32_16x16x32_f16(false, qa[1], false, bk1,
                                                       (short)0, c, false, false);
            sc[n] = c;
        }

        // ---- scale + mask (column = lane&15, uniform across the v8f) ----
        const int kcol = t * KT;
#pragma unroll
        for (int n = 0; n < 4; ++n) {
            const bool masked = (kcol + n * 16 + ln) >= vlen;
#pragma unroll
            for (int v = 0; v < 8; ++v) {
                const float s = sc[n][v] * 0.125f;   // 1/sqrt(64)
                sc[n][v] = masked ? -1.0e6f : s;
            }
        }

        // ---- streaming softmax: row max/sum via xor-shuffles over 16 lanes ----
#pragma unroll
        for (int v = 0; v < 8; ++v) {
            float x = fmaxf(fmaxf(sc[0][v], sc[1][v]), fmaxf(sc[2][v], sc[3][v]));
            x = fmaxf(x, __shfl_xor(x, 1, 32));
            x = fmaxf(x, __shfl_xor(x, 2, 32));
            x = fmaxf(x, __shfl_xor(x, 4, 32));
            x = fmaxf(x, __shfl_xor(x, 8, 32));
            const float mnew  = fmaxf(mrow[v], x);
            const float alpha = __expf(mrow[v] - mnew);  // first iter: exp(-inf)=0
            mrow[v] = mnew;
            lrow[v] *= alpha;
#pragma unroll
            for (int c = 0; c < 4; ++c) oacc[c][v] *= alpha;

            float rs = 0.0f;
#pragma unroll
            for (int n = 0; n < 4; ++n) {
                const float p = __expf(sc[n][v] - mnew);
                sc[n][v] = p;
                rs += p;
            }
            rs += __shfl_xor(rs, 1, 32);
            rs += __shfl_xor(rs, 2, 32);
            rs += __shfl_xor(rs, 4, 32);
            rs += __shfl_xor(rs, 8, 32);
            lrow[v] += rs;
        }

        // ---- P (C layout) -> LDS -> A fragments (transpose within the wave) ----
        _Float16* pw = &Plds[wave][0];
#pragma unroll
        for (int n = 0; n < 4; ++n)
#pragma unroll
            for (int v = 0; v < 8; ++v)
                pw[(v + 8 * kh) * KT + n * 16 + ln] = (_Float16)sc[n][v];

        asm volatile("s_wait_dscnt 0" ::: "memory");   // LDS in-order; keep compiler honest

        v16h pa[2];
#pragma unroll
        for (int c = 0; c < 2; ++c) {
            union { v16h h; unsigned u[8]; } tmp;
#pragma unroll
            for (int v = 0; v < 8; ++v) {
                const int kb = a_kbase(v, kh) + 32 * c;   // even -> dword aligned
                tmp.u[v] = *(const unsigned*)&pw[ln * KT + kb];
            }
            pa[c] = tmp.h;
        }

        // ---- O += P * V  (8 WMMAs; B from transposed V, contiguous reads) ----
#pragma unroll
        for (int c = 0; c < 4; ++c) {
            const v16h bv0 = *(const v16h*)&VTlds[(c * 16 + ln) * KT + 16 * kh];
            const v16h bv1 = *(const v16h*)&VTlds[(c * 16 + ln) * KT + 32 + 16 * kh];
            oacc[c] = __builtin_amdgcn_wmma_f32_16x16x32_f16(false, pa[0], false, bv0,
                                                             (short)0, oacc[c], false, false);
            oacc[c] = __builtin_amdgcn_wmma_f32_16x16x32_f16(false, pa[1], false, bv1,
                                                             (short)0, oacc[c], false, false);
        }
    }

    // ---- epilogue: O / l, fp32 store ----
    const size_t orow = bOff + (size_t)qbase * DIM;
#pragma unroll
    for (int c = 0; c < 4; ++c) {
#pragma unroll
        for (int v = 0; v < 8; ++v) {
            const int m = v + 8 * kh;
            const int d = c * 16 + ln;
            Out[orow + (size_t)m * DIM + d] = oacc[c][v] / lrow[v];
        }
    }
}

extern "C" void kernel_launch(void* const* d_in, const int* in_sizes, int n_in,
                              void* d_out, int out_size, void* d_ws, size_t ws_size,
                              hipStream_t stream) {
    (void)in_sizes; (void)n_in; (void)out_size; (void)d_ws; (void)ws_size;
    const float* Q  = (const float*)d_in[0];
    const float* Km = (const float*)d_in[1];
    const float* Vm = (const float*)d_in[2];
    const int*   VL = (const int*)d_in[3];
    float* Out = (float*)d_out;

    // B=8 batches * 32 query-blocks (128 queries each); 256 threads = 8 waves
    dim3 grid(8 * 32), block(256);
    hipLaunchKernelGGL(fa_masked_sdpa_kernel, grid, block, 0, stream, Q, Km, Vm, VL, Out);
}
